// WindowAttention_21311627723013
// MI455X (gfx1250) — matrix operand order repaired and checked by reading, not compiled, not run on hardware
//
#include <hip/hip_runtime.h>
#include <hip/hip_bf16.h>
#include <stdint.h>

// ---------------------------------------------------------------------------
// Problem constants (from reference)
// ---------------------------------------------------------------------------
#define DIMC   512
#define HEADS  16
#define HD     32
#define NSEQ   256          // post-stride sequence length
#define BWIN   256          // B_ = windows * batch
#define NW     64
#define BH     (BWIN*HEADS) // 4096
#define SCALE  0.17677669529663687f   // 32^-0.5

typedef __attribute__((ext_vector_type(16))) __bf16       v16bf;
typedef __attribute__((ext_vector_type(8)))  float        v8f;
typedef __attribute__((ext_vector_type(4)))  float        v4f;
typedef __attribute__((ext_vector_type(8)))  unsigned int v8u;

union Op16 { v8u u; v16bf b; };

__device__ __forceinline__ v8f zero8() {
  v8f z = {0.f,0.f,0.f,0.f,0.f,0.f,0.f,0.f};
  return z;
}

// f32 -> packed bf16: native CDNA5 V_CVT_PK_BF16_F32 (1 VALU per 2 elems).
// The clang builtin is not declared for gfx1250, so emit it via inline asm.
__device__ __forceinline__ unsigned int pack2bf(float lo, float hi) {
#if defined(__gfx1250__)
  unsigned int r;
  asm("v_cvt_pk_bf16_f32 %0, %1, %2" : "=v"(r) : "v"(lo), "v"(hi));
  return r;
#else
  unsigned int ul = __float_as_uint(lo), uh = __float_as_uint(hi);
  ul = (ul + 0x7FFFu + ((ul >> 16) & 1u)) >> 16;
  uh = (uh + 0x7FFFu + ((uh >> 16) & 1u)) >> 16;
  return (ul & 0xFFFFu) | (uh << 16);
#endif
}
__device__ __forceinline__ unsigned short f2bf(float f) {
  return (unsigned short)pack2bf(f, f);
}

// ---------------------------------------------------------------------------
// 1) Convert weights f32 -> bf16 (same row-major (out, in) layout)
// ---------------------------------------------------------------------------
__global__ __launch_bounds__(256) void wprep_kernel(
    const float* __restrict__ qw, const float* __restrict__ kvw,
    const float* __restrict__ pw,
    unsigned short* __restrict__ wq, unsigned short* __restrict__ wkv,
    unsigned short* __restrict__ wp)
{
  unsigned i = blockIdx.x * 256u + threadIdx.x;    // grid covers 524288
  if (i < DIMC*DIMC)   wq[i]  = f2bf(qw[i]);
  if (i < 2u*DIMC*DIMC) wkv[i] = f2bf(kvw[i]);
  if (i < DIMC*DIMC)   wp[i]  = f2bf(pw[i]);
}

// ---------------------------------------------------------------------------
// 2a) Pre-gather relative position bias, TRANSPOSED for the attention
//     epilogue: rpbP[h][m][n] = table[idx[n][m]][h]
// ---------------------------------------------------------------------------
__global__ __launch_bounds__(256) void rpbprep_kernel(
    const float* __restrict__ table, const int* __restrict__ idx,
    float* __restrict__ rpbP)
{
  unsigned i = blockIdx.x * 256u + threadIdx.x;    // 0 .. 65535  = m*256+n
  unsigned m = i >> 8, n = i & 255u;
  unsigned id = (unsigned)idx[n * 256u + m];
  #pragma unroll
  for (unsigned h = 0; h < HEADS; h++)
    rpbP[h * 65536u + i] = table[id * HEADS + h];
}

// ---------------------------------------------------------------------------
// 2b) Permute window mask to [w][m][n] so epilogue loads are contiguous
// ---------------------------------------------------------------------------
__global__ __launch_bounds__(256) void maskprep_kernel(
    const float* __restrict__ mask, float* __restrict__ maskP)
{
  unsigned i = blockIdx.x * 256u + threadIdx.x;    // covers 64*65536
  unsigned w = i >> 16, r = i & 65535u;
  unsigned m = r >> 8, n = r & 255u;
  maskP[i] = mask[(w << 16) | (n * 256u + m)];
}

// ---------------------------------------------------------------------------
// WMMA operand index helpers (16x16x32 bf16; contiguous k order used by this file)
//   A (16x32): lane&15 = M ; pair base K = 16*hl + 2*p
//   B (32x16): lane&15 = N ; pair base K = 16*hl + 2*p
//   C (16x16): vgpr r -> rows r (lanes 0-15) / r+8 (lanes 16-31), lane&15 = N
// ---------------------------------------------------------------------------

// ---------------------------------------------------------------------------
// 3) Q projection: q = x[:, ::2] @ q_w^T + q_b  -> qbuf bf16 (bh, n, d)
// ---------------------------------------------------------------------------
__global__ __launch_bounds__(256) void qproj_kernel(
    const float* __restrict__ x, const unsigned int* __restrict__ wq,
    const float* __restrict__ qb, unsigned short* __restrict__ qbuf)
{
  const unsigned lane = threadIdx.x & 31u, wv = threadIdx.x >> 5;
  const unsigned hl = lane >> 4, l16 = lane & 15u;
  const unsigned Mband   = (blockIdx.x * 8u + wv) * 16u;  // 512 blocks * 128 rows
  const unsigned colbase = blockIdx.y * 64u;              // 8 blocks * 64 cols

  const unsigned rgA = Mband + l16;                       // A-row this lane feeds
  const unsigned bA = rgA >> 8, nA = rgA & 255u;
  const float* arow = x + ((size_t)bA * 512u + 2u * nA) * DIMC;  // stride-2 row

  v8f acc[4];
  #pragma unroll
  for (int ct = 0; ct < 4; ct++) acc[ct] = zero8();

  for (unsigned kk = 0; kk < 16; kk++) {
    Op16 A;
    #pragma unroll
    for (unsigned p = 0; p < 8; p++) {
      unsigned K = kk * 32u + p * 2u + hl * 16u;
      A.u[p] = pack2bf(arow[K], arow[K + 1]);
    }
    #pragma unroll
    for (unsigned ct = 0; ct < 4; ct++) {
      unsigned o = colbase + ct * 16u + l16;
      const unsigned int* wrow = wq + (size_t)o * 256u + kk * 16u + hl * 8u;
      Op16 B;
      #pragma unroll
      for (unsigned p = 0; p < 8; p++) B.u[p] = wrow[p];
      acc[ct] = __builtin_amdgcn_wmma_f32_16x16x32_bf16(
          false, A.b, false, B.b, (short)0, acc[ct], false, false);
    }
  }
  #pragma unroll
  for (unsigned ct = 0; ct < 4; ct++) {
    unsigned o = colbase + ct * 16u + l16;
    float bias = qb[o];
    unsigned h = o >> 5, d = o & 31u;
    #pragma unroll
    for (unsigned r = 0; r < 8; r++) {
      unsigned rg = Mband + r + hl * 8u;
      unsigned b2 = rg >> 8, n2 = rg & 255u;
      qbuf[(((size_t)b2 * HEADS + h) * NSEQ + n2) * HD + d] =
          f2bf(acc[ct][r] + bias);
    }
  }
}

// ---------------------------------------------------------------------------
// 4) KV projection: kv = y[:, ::2] @ kv_w^T + kv_b
//    k -> kbuf bf16 (bh, m, d) ; v -> vTbuf bf16 (bh, d, m)  (pre-transposed)
// ---------------------------------------------------------------------------
__global__ __launch_bounds__(256) void kvproj_kernel(
    const float* __restrict__ y, const unsigned int* __restrict__ wkv,
    const float* __restrict__ kvb, unsigned short* __restrict__ kbuf,
    unsigned short* __restrict__ vTbuf)
{
  const unsigned lane = threadIdx.x & 31u, wv = threadIdx.x >> 5;
  const unsigned hl = lane >> 4, l16 = lane & 15u;
  const unsigned Mband   = (blockIdx.x * 8u + wv) * 16u;
  const unsigned colbase = blockIdx.y * 64u;              // 16 blocks * 64 cols

  const unsigned rgA = Mband + l16;
  const unsigned bA = rgA >> 8, nA = rgA & 255u;
  const float* arow = y + ((size_t)bA * 512u + 2u * nA) * DIMC;

  v8f acc[4];
  #pragma unroll
  for (int ct = 0; ct < 4; ct++) acc[ct] = zero8();

  for (unsigned kk = 0; kk < 16; kk++) {
    Op16 A;
    #pragma unroll
    for (unsigned p = 0; p < 8; p++) {
      unsigned K = kk * 32u + p * 2u + hl * 16u;
      A.u[p] = pack2bf(arow[K], arow[K + 1]);
    }
    #pragma unroll
    for (unsigned ct = 0; ct < 4; ct++) {
      unsigned o = colbase + ct * 16u + l16;
      const unsigned int* wrow = wkv + (size_t)o * 256u + kk * 16u + hl * 8u;
      Op16 B;
      #pragma unroll
      for (unsigned p = 0; p < 8; p++) B.u[p] = wrow[p];
      acc[ct] = __builtin_amdgcn_wmma_f32_16x16x32_bf16(
          false, A.b, false, B.b, (short)0, acc[ct], false, false);
    }
  }
  #pragma unroll
  for (unsigned ct = 0; ct < 4; ct++) {
    unsigned o = colbase + ct * 16u + l16;
    float bias = kvb[o];
    if (o < DIMC) {                                 // K half
      unsigned h = o >> 5, d = o & 31u;
      #pragma unroll
      for (unsigned r = 0; r < 8; r++) {
        unsigned rg = Mband + r + hl * 8u;
        unsigned b2 = rg >> 8, n2 = rg & 255u;
        kbuf[(((size_t)b2 * HEADS + h) * NSEQ + n2) * HD + d] =
            f2bf(acc[ct][r] + bias);
      }
    } else {                                        // V half, store transposed
      unsigned o2 = o - DIMC;
      unsigned h = o2 >> 5, d = o2 & 31u;
      #pragma unroll
      for (unsigned r = 0; r < 8; r++) {
        unsigned rg = Mband + r + hl * 8u;
        unsigned b2 = rg >> 8, n2 = rg & 255u;
        vTbuf[(((size_t)b2 * HEADS + h) * HD + d) * NSEQ + n2] =
            f2bf(acc[ct][r] + bias);
      }
    }
  }
}

// ---------------------------------------------------------------------------
// 5) Attention: one workgroup (4 waves) per (b,h). Wave owns 64 rows,
//    processed as four 16-row bands held entirely in accumulators.
// ---------------------------------------------------------------------------
__global__ __launch_bounds__(128) void attn_kernel(
    const unsigned int* __restrict__ qbuf, const unsigned int* __restrict__ kbuf,
    const unsigned int* __restrict__ vTbuf, const float* __restrict__ rpbP,
    const float* __restrict__ maskP, unsigned short* __restrict__ obuf)
{
  __shared__ unsigned short p_sh[4][16 * 256];      // 8 KB per wave, 32 KB total

  const unsigned bh = blockIdx.x;
  const unsigned b  = bh >> 4, h = bh & 15u;
  const unsigned lane = threadIdx.x & 31u, wv = threadIdx.x >> 5;
  const unsigned hl = lane >> 4, l16 = lane & 15u;

  const unsigned int* qg = qbuf  + (size_t)bh * 4096u;  // 256*32 bf16 = 4096 dw
  const unsigned int* kg = kbuf  + (size_t)bh * 4096u;
  const unsigned int* vg = vTbuf + (size_t)bh * 4096u;
  const float* rpbh = rpbP  + (size_t)h * 65536u;                 // [m][n]
  const float* mw   = maskP + (size_t)(b & (NW - 1u)) * 65536u;   // [m][n]
  unsigned short* ps = p_sh[wv];

  for (unsigned band = 0; band < 4; band++) {
    const unsigned rowbase = wv * 64u + band * 16u;

    // ---- S = q * k^T  (16 x 256, K = 32) --------------------------------
    Op16 Aq;
    #pragma unroll
    for (unsigned p = 0; p < 8; p++)
      Aq.u[p] = qg[(rowbase + l16) * 16u + p + hl * 8u];

    v8f acc[16];
    #pragma unroll
    for (int ct = 0; ct < 16; ct++) acc[ct] = zero8();
    #pragma unroll
    for (unsigned ct = 0; ct < 16; ct++) {
      unsigned m = ct * 16u + l16;                  // B: N = m column
      const unsigned int* kr = kg + m * 16u + hl * 8u;
      Op16 Bk;
      #pragma unroll
      for (unsigned p = 0; p < 8; p++) Bk.u[p] = kr[p];
      acc[ct] = __builtin_amdgcn_wmma_f32_16x16x32_bf16(
          false, Aq.b, false, Bk.b, (short)0, acc[ct], false, false);
    }

    // ---- scale + rel-pos bias + window mask, row softmax ----------------
    // rpbP/maskP are [m][n]: this lane's 8 rows (n) are contiguous -> float4
    float mx[8], sm[8];
    #pragma unroll
    for (int r = 0; r < 8; r++) mx[r] = -1e30f;
    #pragma unroll
    for (unsigned ct = 0; ct < 16; ct++) {
      unsigned m = ct * 16u + l16;
      const float* rr = rpbh + m * 256u + rowbase + hl * 8u;
      const float* mm = mw   + m * 256u + rowbase + hl * 8u;
      v4f ra = *(const v4f*)rr;
      v4f rb = *(const v4f*)(rr + 4);
      v4f ma = *(const v4f*)mm;
      v4f mb = *(const v4f*)(mm + 4);
      #pragma unroll
      for (int r = 0; r < 4; r++) {
        float v = acc[ct][r] * SCALE + ra[r] + ma[r];
        acc[ct][r] = v;
        mx[r] = fmaxf(mx[r], v);
      }
      #pragma unroll
      for (int r = 4; r < 8; r++) {
        float v = acc[ct][r] * SCALE + rb[r - 4] + mb[r - 4];
        acc[ct][r] = v;
        mx[r] = fmaxf(mx[r], v);
      }
    }
    #pragma unroll
    for (int r = 0; r < 8; r++) {                   // reduce within 16-lane group
      float v = mx[r];
      v = fmaxf(v, __shfl_xor(v, 1));
      v = fmaxf(v, __shfl_xor(v, 2));
      v = fmaxf(v, __shfl_xor(v, 4));
      v = fmaxf(v, __shfl_xor(v, 8));
      mx[r] = v; sm[r] = 0.f;
    }
    #pragma unroll
    for (int ct = 0; ct < 16; ct++) {
      #pragma unroll
      for (int r = 0; r < 8; r++) {
        float e = __expf(acc[ct][r] - mx[r]);
        acc[ct][r] = e; sm[r] += e;
      }
    }
    #pragma unroll
    for (int r = 0; r < 8; r++) {
      float s = sm[r];
      s += __shfl_xor(s, 1); s += __shfl_xor(s, 2);
      s += __shfl_xor(s, 4); s += __shfl_xor(s, 8);
      sm[r] = 1.0f / s;
    }

    // ---- P (C-layout) -> LDS row-major 16x256 bf16 ----------------------
    #pragma unroll
    for (unsigned ct = 0; ct < 16; ct++) {
      unsigned m = ct * 16u + l16;
      #pragma unroll
      for (unsigned r = 0; r < 8; r++) {
        unsigned nl = r + hl * 8u;
        ps[nl * 256u + m] = f2bf(acc[ct][r] * sm[r]);
      }
    }
    __syncthreads();

    // ---- O = P * V  (16 x 32, K = 256 in 8 chunks) ----------------------
    const unsigned int* pd = (const unsigned int*)ps;
    v8f accO[2];
    accO[0] = zero8(); accO[1] = zero8();
    #pragma unroll
    for (unsigned kk = 0; kk < 8; kk++) {
      Op16 Ap;
      #pragma unroll
      for (unsigned p = 0; p < 8; p++)
        Ap.u[p] = pd[l16 * 128u + kk * 16u + p + hl * 8u];
      #pragma unroll
      for (unsigned ct2 = 0; ct2 < 2; ct2++) {
        unsigned d = ct2 * 16u + l16;               // B: N = d column of vT
        const unsigned int* vr = vg + d * 128u + kk * 16u + hl * 8u;
        Op16 Bv;
        #pragma unroll
        for (unsigned p = 0; p < 8; p++) Bv.u[p] = vr[p];
        accO[ct2] = __builtin_amdgcn_wmma_f32_16x16x32_bf16(
            false, Ap.b, false, Bv.b, (short)0, accO[ct2], false, false);
      }
    }

    // ---- store O into (b*256+n, h*32+d) bf16 for the projection GEMM ----
    #pragma unroll
    for (unsigned ct2 = 0; ct2 < 2; ct2++) {
      unsigned d = ct2 * 16u + l16;
      #pragma unroll
      for (unsigned r = 0; r < 8; r++) {
        unsigned n = rowbase + r + hl * 8u;
        obuf[((size_t)b * NSEQ + n) * DIMC + h * HD + d] = f2bf(accO[ct2][r]);
      }
    }
  }
}

// ---------------------------------------------------------------------------
// 6) Output projection: out = O @ proj_w^T + proj_b (f32 result)
// ---------------------------------------------------------------------------
__global__ __launch_bounds__(256) void proj_kernel(
    const unsigned int* __restrict__ ob, const unsigned int* __restrict__ wp,
    const float* __restrict__ pb, float* __restrict__ out)
{
  const unsigned lane = threadIdx.x & 31u, wv = threadIdx.x >> 5;
  const unsigned hl = lane >> 4, l16 = lane & 15u;
  const unsigned Mband   = (blockIdx.x * 8u + wv) * 16u;
  const unsigned colbase = blockIdx.y * 64u;

  const unsigned int* arow = ob + (size_t)(Mband + l16) * 256u; // bf16 row, dwords

  v8f acc[4];
  #pragma unroll
  for (int ct = 0; ct < 4; ct++) acc[ct] = zero8();

  for (unsigned kk = 0; kk < 16; kk++) {
    Op16 A;
    #pragma unroll
    for (unsigned p = 0; p < 8; p++)
      A.u[p] = arow[kk * 16u + p + hl * 8u];
    #pragma unroll
    for (unsigned ct = 0; ct < 4; ct++) {
      unsigned o = colbase + ct * 16u + l16;
      const unsigned int* wrow = wp + (size_t)o * 256u + kk * 16u + hl * 8u;
      Op16 B;
      #pragma unroll
      for (unsigned p = 0; p < 8; p++) B.u[p] = wrow[p];
      acc[ct] = __builtin_amdgcn_wmma_f32_16x16x32_bf16(
          false, A.b, false, B.b, (short)0, acc[ct], false, false);
    }
  }
  #pragma unroll
  for (unsigned ct = 0; ct < 4; ct++) {
    unsigned o = colbase + ct * 16u + l16;
    float bias = pb[o];
    #pragma unroll
    for (unsigned r = 0; r < 8; r++)
      out[(size_t)(Mband + r + hl * 8u) * DIMC + o] = acc[ct][r] + bias;
  }
}

// ---------------------------------------------------------------------------
// Host-side launcher
// ---------------------------------------------------------------------------
extern "C" void kernel_launch(void* const* d_in, const int* in_sizes, int n_in,
                              void* d_out, int out_size, void* d_ws, size_t ws_size,
                              hipStream_t stream)
{
  const float* x     = (const float*)d_in[0];
  const float* y     = (const float*)d_in[1];
  const float* mask  = (const float*)d_in[2];
  const float* q_w   = (const float*)d_in[3];
  const float* q_b   = (const float*)d_in[4];
  const float* kv_w  = (const float*)d_in[5];
  const float* kv_b  = (const float*)d_in[6];
  const float* p_w   = (const float*)d_in[7];
  const float* p_b   = (const float*)d_in[8];
  const float* btab  = (const float*)d_in[9];
  const int*   rpi   = (const int*)d_in[10];
  (void)in_sizes; (void)n_in; (void)out_size; (void)ws_size;

  char*  ws  = (char*)d_ws;
  size_t off = 0;
  auto alloc = [&](size_t bytes) -> void* {
    void* p = ws + off;
    off += (bytes + 255) & ~(size_t)255;
    return p;
  };
  unsigned short* wq    = (unsigned short*)alloc((size_t)DIMC * DIMC * 2);
  unsigned short* wkv   = (unsigned short*)alloc((size_t)2 * DIMC * DIMC * 2);
  unsigned short* wp    = (unsigned short*)alloc((size_t)DIMC * DIMC * 2);
  float*          rpbP  = (float*)alloc((size_t)HEADS * 65536 * 4);
  float*          maskP = (float*)alloc((size_t)NW * 65536 * 4);
  unsigned short* qbuf  = (unsigned short*)alloc((size_t)BH * NSEQ * HD * 2);
  unsigned short* kbuf  = (unsigned short*)alloc((size_t)BH * NSEQ * HD * 2);
  unsigned short* vTbuf = (unsigned short*)alloc((size_t)BH * NSEQ * HD * 2);
  unsigned short* obuf  = (unsigned short*)alloc((size_t)BWIN * NSEQ * DIMC * 2);

  wprep_kernel<<<2048, 256, 0, stream>>>(q_w, kv_w, p_w, wq, wkv, wp);
  rpbprep_kernel<<<256, 256, 0, stream>>>(btab, rpi, rpbP);
  maskprep_kernel<<<16384, 256, 0, stream>>>(mask, maskP);
  qproj_kernel<<<dim3(512, 8), 256, 0, stream>>>(
      x, (const unsigned int*)wq, q_b, qbuf);
  kvproj_kernel<<<dim3(512, 16), 256, 0, stream>>>(
      y, (const unsigned int*)wkv, kv_b, kbuf, vTbuf);
  attn_kernel<<<BH, 128, 0, stream>>>(
      (const unsigned int*)qbuf, (const unsigned int*)kbuf,
      (const unsigned int*)vTbuf, rpbP, maskP, obuf);
  proj_kernel<<<dim3(512, 8), 256, 0, stream>>>(
      (const unsigned int*)obuf, (const unsigned int*)wp, p_b, (float*)d_out);
}
